// SampleFlashAttention_28346784154102
// MI455X (gfx1250) — matrix operand order repaired and checked by reading, not compiled, not run on hardware
//
#include <hip/hip_runtime.h>

typedef __attribute__((ext_vector_type(16))) __bf16 v16bf;
typedef __attribute__((ext_vector_type(8)))  __bf16 v8bf;
typedef __attribute__((ext_vector_type(8)))  float  v8f;

#define NSAMP  256
#define LPOS   512
#define DMODEL 256
#define NHEAD  8
#define HD     32

#if defined(__AMDGCN__)
#define ATTN_USE_TDM 1
typedef __attribute__((ext_vector_type(4))) unsigned int tdm_u4;
typedef __attribute__((ext_vector_type(8))) int          tdm_i8;
#else
#define ATTN_USE_TDM 0
#endif

// ---- fragment helpers -------------------------------------------------------

// 16 contiguous bf16 split as two 8-element halves (A low/high K octets or B rows)
static __device__ __forceinline__ v16bf ld_bf16_pair(const __bf16* lo, const __bf16* hi) {
  v8bf a = *(const v8bf*)lo;
  v8bf b = *(const v8bf*)hi;
  return __builtin_shufflevector(a, b, 0,1,2,3,4,5,6,7,8,9,10,11,12,13,14,15);
}

// A-fragment from fp32 row: halves 0..7 = K kbase..kbase+7, halves 8..15 = K kbase+8..kbase+15
static __device__ __forceinline__ v16bf cvt_a_from_f32(const float* row, int kbase) {
  float4 a = *(const float4*)(row + kbase);
  float4 b = *(const float4*)(row + kbase + 4);
  float4 c = *(const float4*)(row + kbase + 8);
  float4 d = *(const float4*)(row + kbase + 12);
  v16bf r;
  r[0]=(__bf16)a.x;  r[1]=(__bf16)a.y;  r[2]=(__bf16)a.z;  r[3]=(__bf16)a.w;
  r[4]=(__bf16)b.x;  r[5]=(__bf16)b.y;  r[6]=(__bf16)b.z;  r[7]=(__bf16)b.w;
  r[8]=(__bf16)c.x;  r[9]=(__bf16)c.y;  r[10]=(__bf16)c.z; r[11]=(__bf16)c.w;
  r[12]=(__bf16)d.x; r[13]=(__bf16)d.y; r[14]=(__bf16)d.z; r[15]=(__bf16)d.w;
  return r;
}

// B-fragment from 16 contiguous fp32 (K contiguous per lane)
static __device__ __forceinline__ v16bf cvt_b_from_f32(const float* p) {
  float4 a = *(const float4*)(p);
  float4 b = *(const float4*)(p + 4);
  float4 c = *(const float4*)(p + 8);
  float4 d = *(const float4*)(p + 12);
  v16bf r;
  r[0]=(__bf16)a.x;  r[1]=(__bf16)a.y;  r[2]=(__bf16)a.z;  r[3]=(__bf16)a.w;
  r[4]=(__bf16)b.x;  r[5]=(__bf16)b.y;  r[6]=(__bf16)b.z;  r[7]=(__bf16)b.w;
  r[8]=(__bf16)c.x;  r[9]=(__bf16)c.y;  r[10]=(__bf16)c.z; r[11]=(__bf16)c.w;
  r[12]=(__bf16)d.x; r[13]=(__bf16)d.y; r[14]=(__bf16)d.z; r[15]=(__bf16)d.w;
  return r;
}

// ---- stage 1: QKV projection, scatter to (h,l,n,hd) bf16 layout -------------
// Each wave: 16(M) x 64(E) tile, K=256.  M = n*L + l over x rows.
__global__ __launch_bounds__(256) void qkv_gemm(
    const float* __restrict__ x, const float* __restrict__ wqkv,
    __bf16* __restrict__ Qb, __bf16* __restrict__ Kb, __bf16* __restrict__ Vb) {
  const int lane = threadIdx.x & 31;
  const int wave = threadIdx.x >> 5;
  const int wt   = blockIdx.x * 8 + wave;     // 8192*12 = 98304 wave tiles
  const int m0   = (wt / 12) * 16;
  const int e0   = (wt % 12) * 64;
  const int hl   = lane & 15;
  const bool hi  = lane >= 16;
  const int akb  = hi ? 16 : 0;

  const float* arow = x + (size_t)(m0 + hl) * DMODEL;
  v8f acc[4] = {};
#pragma unroll
  for (int k0 = 0; k0 < DMODEL; k0 += 32) {
    v16bf A = cvt_a_from_f32(arow, k0 + akb);
#pragma unroll
    for (int s = 0; s < 4; ++s) {
      int e = e0 + s * 16 + hl;
      v16bf B = cvt_b_from_f32(wqkv + (size_t)e * DMODEL + k0 + (hi ? 16 : 0));
      acc[s] = __builtin_amdgcn_wmma_f32_16x16x32_bf16(
          false, A, false, B, (short)0, acc[s], false, false);
    }
  }
  const float qscale = 0.17677669529663687f;  // hd^-0.5
#pragma unroll
  for (int s = 0; s < 4; ++s) {
    int e = e0 + s * 16 + hl;
    int t = e >> 8;                // 0=q 1=k 2=v (uniform per s)
    int d = e & 255;
    int h = d >> 5, c = d & 31;
    __bf16* dst = (t == 0) ? Qb : (t == 1 ? Kb : Vb);
#pragma unroll
    for (int r = 0; r < 8; ++r) {
      int m = m0 + r + (hi ? 8 : 0);
      int n = m >> 9;              // L = 512
      int l = m & 511;
      float v = acc[s][r];
      if (t == 0) v *= qscale;
      dst[(((size_t)h * LPOS + l) * NSAMP + n) * HD + c] = (__bf16)v;
    }
  }
}

// ---- stage 2: attention over sample axis per (h,l) --------------------------
// Block = 128 thr (4 waves), covers 64 query rows. grid = (4096, 4).
__global__ __launch_bounds__(128) void attn_kernel(
    const __bf16* __restrict__ Qb, const __bf16* __restrict__ Kb,
    const __bf16* __restrict__ Vb, const float* __restrict__ bias,
    __bf16* __restrict__ Ob) {
  __shared__ __align__(16) unsigned char smem[49152];
  __bf16* Vt = (__bf16*)smem;             // Vt[c*256 + m] (transposed V), 16 KB
  __bf16* Ks = (__bf16*)(smem + 16384);   // Ks[n*32 + k], 16 KB (dead after scores)

  const int tid = threadIdx.x, lane = tid & 31, wave = tid >> 5;
  const int hl = lane & 15;
  const bool hi = lane >= 16;
  const int akb = hi ? 16 : 0;
  const int hlid = blockIdx.x;            // h*512 + l
  const int h = hlid >> 9, l = hlid & 511;
  const size_t mat = (size_t)hlid * (NSAMP * HD);

#if ATTN_USE_TDM
  // K tile (16 KB contiguous) via Tensor Data Mover, issued from wave 0 and
  // overlapped with the VALU/LDS V-transpose below.  Inline asm (2-group form,
  // VADDR2/3 = NULL -> <=2D tensor) is portable across both toolchains.
  if (wave == 0) {
    const unsigned long long ga = (unsigned long long)(const void*)(Kb + mat);
    const unsigned lds_off = (unsigned)(unsigned long long)(const void*)Ks;
    tdm_u4 g0;
    g0[0] = 1u;                                               // count=1 (valid user D#)
    g0[1] = lds_off;                                          // lds_addr (bytes)
    g0[2] = (unsigned)(ga & 0xFFFFFFFFu);                     // global_addr[31:0]
    g0[3] = (unsigned)((ga >> 32) & 0x1FFFFFFu) | (2u << 30); // addr[56:32], type=2
    tdm_i8 g1;
    g1[0] = (int)(1u << 16);        // wg_mask=0, data_size=1 (2-byte elems)
    g1[1] = (int)(8192u << 16);     // tensor_dim0 = 8192 elements
    g1[2] = (int)(1u << 16);        // tensor_dim0 hi=0, tensor_dim1 = 1
    g1[3] = (int)(8192u << 16);     // tensor_dim1 hi=0, tile_dim0 = 8192
    g1[4] = 1;                      // tile_dim1 = 1, tile_dim2 = 0
    g1[5] = 8192;                   // tensor_dim0_stride lo
    g1[6] = (int)(8192u << 16);     // stride0 hi=0, tensor_dim1_stride lo16
    g1[7] = 0;
    asm volatile("tensor_load_to_lds %0, %1" :: "s"(g0), "s"(g1) : "memory");
  }
#else
  for (int i4 = tid; i4 < 2048; i4 += 128)
    ((uint2*)Ks)[i4] = ((const uint2*)(Kb + mat))[i4];
#endif

  // cooperative V transpose: global -> regs -> LDS scatter
  for (int i4 = tid; i4 < 2048; i4 += 128) {
    int m = i4 >> 3, c4 = (i4 & 7) * 4;
    uint2 raw = ((const uint2*)(Vb + mat))[i4];
    __bf16 tmp[4]; *(uint2*)tmp = raw;
#pragma unroll
    for (int i = 0; i < 4; ++i) Vt[(c4 + i) * NSAMP + m] = tmp[i];
  }
#if ATTN_USE_TDM
  if (wave == 0) asm volatile("s_wait_tensorcnt 0x0" ::: "memory");
#endif
  __syncthreads();

  const int rowbase = blockIdx.y * 64 + wave * 16;
  // bias rows for this wave live in L2 (256 KB array, reused 4096x): prefetch
  __builtin_prefetch(bias + (size_t)(rowbase + hl) * NSAMP, 0, 1);

  const __bf16* qrow = Qb + mat + (size_t)(rowbase + hl) * HD;
  v16bf Aq = ld_bf16_pair(qrow + akb, qrow + akb + 8);

  // scores: 16 column tiles of 16, K-dim = hd = 32 (one WMMA each)
  v8f acc[16] = {};
#pragma unroll
  for (int j = 0; j < 16; ++j) {
    const __bf16* kp = Ks + (j * 16 + hl) * HD + (hi ? 16 : 0);
    v16bf Bf = ld_bf16_pair(kp, kp + 8);
    acc[j] = __builtin_amdgcn_wmma_f32_16x16x32_bf16(
        false, Aq, false, Bf, (short)0, acc[j], false, false);
  }

  // bias + row softmax (rows r and r+8 live in lane halves; reduce across 16 lanes)
  float rmax[8], rsum[8], rinv[8];
#pragma unroll
  for (int r = 0; r < 8; ++r) rmax[r] = -3.0e38f;
  const int growb = rowbase + (hi ? 8 : 0);
#pragma unroll
  for (int j = 0; j < 16; ++j) {
    int gcol = j * 16 + hl;
#pragma unroll
    for (int r = 0; r < 8; ++r) {
      float v = acc[j][r] + bias[(growb + r) * NSAMP + gcol];
      acc[j][r] = v;
      rmax[r] = fmaxf(rmax[r], v);
    }
  }
#pragma unroll
  for (int r = 0; r < 8; ++r)
#pragma unroll
    for (int m = 1; m < 16; m <<= 1)
      rmax[r] = fmaxf(rmax[r], __shfl_xor(rmax[r], m, 32));
#pragma unroll
  for (int r = 0; r < 8; ++r) rsum[r] = 0.f;
#pragma unroll
  for (int j = 0; j < 16; ++j)
#pragma unroll
    for (int r = 0; r < 8; ++r) {
      float e = __expf(acc[j][r] - rmax[r]);
      acc[j][r] = e;
      rsum[r] += e;
    }
#pragma unroll
  for (int r = 0; r < 8; ++r) {
#pragma unroll
    for (int m = 1; m < 16; m <<= 1)
      rsum[r] += __shfl_xor(rsum[r], m, 32);
    rinv[r] = 1.0f / rsum[r];
  }

  // re-fragment exp(scores) via LDS (overlay on Ks region; Ks is dead now)
  __syncthreads();
  __bf16* stg = (__bf16*)(smem + 16384 + wave * 8192);  // [16][256] per wave
#pragma unroll
  for (int j = 0; j < 16; ++j)
#pragma unroll
    for (int r = 0; r < 8; ++r)
      stg[(r + (hi ? 8 : 0)) * NSAMP + j * 16 + hl] = (__bf16)acc[j][r];

  // out = exp(scores) @ V : 16 x 32, K = 256 in 8 steps of 32
  v8f o[2] = {};
#pragma unroll
  for (int kk = 0; kk < 8; ++kk) {
    const __bf16* ap = stg + hl * NSAMP + kk * 32 + akb;
    v16bf Af = ld_bf16_pair(ap, ap + 8);
#pragma unroll
    for (int ct = 0; ct < 2; ++ct) {
      const __bf16* bp = Vt + (ct * 16 + hl) * NSAMP + kk * 32 + (hi ? 16 : 0);
      v16bf Bf = ld_bf16_pair(bp, bp + 8);
      o[ct] = __builtin_amdgcn_wmma_f32_16x16x32_bf16(
          false, Af, false, Bf, (short)0, o[ct], false, false);
    }
  }

  // normalize and scatter to (n,l,d) bf16
#pragma unroll
  for (int ct = 0; ct < 2; ++ct)
#pragma unroll
    for (int r = 0; r < 8; ++r) {
      int n = rowbase + r + (hi ? 8 : 0);
      int c = ct * 16 + hl;
      float v = o[ct][r] * rinv[r];
      Ob[((size_t)n * LPOS + l) * DMODEL + h * HD + c] = (__bf16)v;
    }
}

// ---- stage 3: output projection + bias, fp32 out ---------------------------
__global__ __launch_bounds__(256) void proj_gemm(
    const __bf16* __restrict__ O, const float* __restrict__ wproj,
    const float* __restrict__ bproj, float* __restrict__ out) {
  const int lane = threadIdx.x & 31;
  const int wave = threadIdx.x >> 5;
  const int wt   = blockIdx.x * 8 + wave;     // 8192*4 = 32768 wave tiles
  const int m0   = (wt >> 2) * 16;
  const int e0   = (wt & 3) * 64;
  const int hl   = lane & 15;
  const bool hi  = lane >= 16;
  const int akb  = hi ? 16 : 0;

  const __bf16* arow = O + (size_t)(m0 + hl) * DMODEL;
  v8f acc[4] = {};
#pragma unroll
  for (int k0 = 0; k0 < DMODEL; k0 += 32) {
    v16bf A = ld_bf16_pair(arow + k0 + akb, arow + k0 + akb + 8);
#pragma unroll
    for (int s = 0; s < 4; ++s) {
      int e = e0 + s * 16 + hl;
      v16bf B = cvt_b_from_f32(wproj + (size_t)e * DMODEL + k0 + (hi ? 16 : 0));
      acc[s] = __builtin_amdgcn_wmma_f32_16x16x32_bf16(
          false, A, false, B, (short)0, acc[s], false, false);
    }
  }
#pragma unroll
  for (int s = 0; s < 4; ++s) {
    int e = e0 + s * 16 + hl;
    float bb = bproj[e];
#pragma unroll
    for (int r = 0; r < 8; ++r) {
      int m = m0 + r + (hi ? 8 : 0);
      out[(size_t)m * DMODEL + e] = acc[s][r] + bb;
    }
  }
}

// ---- launch -----------------------------------------------------------------
extern "C" void kernel_launch(void* const* d_in, const int* in_sizes, int n_in,
                              void* d_out, int out_size, void* d_ws, size_t ws_size,
                              hipStream_t stream) {
  const float* x     = (const float*)d_in[0];
  const float* bias  = (const float*)d_in[1];
  const float* wqkv  = (const float*)d_in[2];
  const float* wproj = (const float*)d_in[3];
  const float* bproj = (const float*)d_in[4];
  float* out = (float*)d_out;

  const size_t elems = (size_t)NSAMP * LPOS * DMODEL;  // 33,554,432
  __bf16* Qb = (__bf16*)d_ws;          // needs 4*elems*2 = 256 MB workspace
  __bf16* Kb = Qb + elems;
  __bf16* Vb = Kb + elems;
  __bf16* Ob = Vb + elems;

  qkv_gemm<<<12288, 256, 0, stream>>>(x, wqkv, Qb, Kb, Vb);
  attn_kernel<<<dim3(4096, 4, 1), 128, 0, stream>>>(Qb, Kb, Vb, bias, Ob);
  proj_gemm<<<4096, 256, 0, stream>>>(Ob, wproj, bproj, out);
}
